// CitGIN_90056874262918
// MI455X (gfx1250) — matrix operand order repaired and checked by reading, not compiled, not run on hardware
//
#include <hip/hip_runtime.h>

// ---------------------------------------------------------------------------
// CitGIN on MI455X (gfx1250): bf16 WMMA GEMMs + fp32 L2 atomic scatter-add.
// Round 2: 80-row M-blocks, 5 output tiles per wave, B-fragment reuse.
// ---------------------------------------------------------------------------

#define GIN_N 50000
#define GIN_E 800000
#define GIN_F 128
#define GIN_H 128
#define GIN_C 40

#define MROWS  80      // node rows per block (50000 / 80 = 625 exact)
#define MTILES 5       // 16-row WMMA subtiles per block

typedef __attribute__((ext_vector_type(16))) __bf16          v16bf;
typedef __attribute__((ext_vector_type(16))) unsigned short  v16u16;
typedef __attribute__((ext_vector_type(8)))  float           v8f;

__device__ __forceinline__ unsigned short f2bf(float f) {
    // round-to-nearest-even fp32 -> bf16 (NaN edge cases ignored)
    unsigned u = __float_as_uint(f);
    u += 0x7FFFu + ((u >> 16) & 1u);
    return (unsigned short)(u >> 16);
}

// ---------------- weight fp32 -> bf16 conversion ---------------------------
__global__ void gin_f2bf(const float* __restrict__ in, unsigned short* __restrict__ out, int n) {
    int i = blockIdx.x * blockDim.x + threadIdx.x;
    if (i < n) out[i] = f2bf(in[i]);
}

// ---------------- float4 copy (agg init) -----------------------------------
__global__ void gin_copy4(const float4* __restrict__ in, float4* __restrict__ out, int n4) {
    int i = blockIdx.x * blockDim.x + threadIdx.x;
    if (i < n4) out[i] = in[i];
}

// ---------------- edge scatter-add: one wave per edge ----------------------
// agg[dst] += h[src]; 32 lanes x 4 floats = 128 features; atomics hit L2.
__global__ __launch_bounds__(256) void gin_scatter(const int* __restrict__ ei,
                                                   const float* __restrict__ hs,
                                                   float* __restrict__ hd) {
    int e    = (blockIdx.x * blockDim.x + threadIdx.x) >> 5;
    int lane = threadIdx.x & 31;
    if (e >= GIN_E) return;
    int s = ei[e];
    int d = ei[GIN_E + e];
    float4 v = ((const float4*)(hs + (size_t)s * GIN_H))[lane];
    float* pd = hd + (size_t)d * GIN_H + lane * 4;
    atomicAdd(pd + 0, v.x);
    atomicAdd(pd + 1, v.y);
    atomicAdd(pd + 2, v.z);
    atomicAdd(pd + 3, v.w);
}

// ---------------- WMMA GEMM: D = act( A[50000x128] * W^T ) -----------------
// W is [NCOLS x 128] row-major (torch Linear layout), so B[k][o] = W[o*128+k].
// Block = 80 node rows staged once in LDS (bf16, 20 KB); one wave per
// 16-column tile; each wave sweeps 5 M-subtiles reusing one B fragment per
// K-step -> 20 v_wmma per wave, 5x less B traffic, 5x amortized epilogue.
template <bool A_BF16, bool DO_BN, bool OUT_BF16, bool DUAL_OUT, int NCOLS>
__global__ __launch_bounds__(256) void gin_gemm(
    const void* __restrict__ Asrc,
    const unsigned short* __restrict__ Wb,
    const float* __restrict__ gamma, const float* __restrict__ beta,
    const float* __restrict__ mean,  const float* __restrict__ var,
    void* __restrict__ Dout, float* __restrict__ Dout2) {

    __shared__ unsigned short As[MROWS * 128];   // 20 KB bf16 A tile

    const int m0  = blockIdx.x * MROWS;
    const int tid = threadIdx.x;

    // cooperative A tile load (convert fp32->bf16 if needed)
    for (int idx = tid; idx < MROWS * 128; idx += blockDim.x) {
        int r = idx >> 7, k = idx & 127;
        size_t g = (size_t)(m0 + r) * 128 + k;
        unsigned short v;
        if (A_BF16) v = ((const unsigned short*)Asrc)[g];
        else        v = f2bf(((const float*)Asrc)[g]);
        As[idx] = v;
    }
    __syncthreads();

    const int wv   = tid >> 5;        // wave id -> column tile
    const int lane = tid & 31;
    const int o0   = wv * 16;
    const int n    = lane & 15;       // column / M-sub row selector
    const int hh   = lane >> 4;       // lane half
    const int ocol = o0 + n;

    v8f c[MTILES];
    #pragma unroll
    for (int t = 0; t < MTILES; ++t) c[t] = {};

    #pragma unroll
    for (int kb = 0; kb < 128; kb += 32) {
        // ---- B fragment (loaded once per K-step, reused by all M tiles) ----
        // 32x16 bf16: VGPR v<4 holds K rows {8*hh+2v, +1}, v>=4 holds {16+8*hh+2(v-4), +1}, column = lane&15.
        v16u16 bu;
        #pragma unroll
        for (int v = 0; v < 8; ++v) {
            int k = kb + (v < 4 ? 8 * hh + 2 * v : 16 + 8 * hh + 2 * (v - 4));
            unsigned short b0 = 0, b1 = 0;
            if (NCOLS != 40 || ocol < GIN_C) {
                b0 = Wb[(size_t)ocol * 128 + k];
                b1 = Wb[(size_t)ocol * 128 + k + 1];
            }
            bu[2 * v]     = b0;
            bu[2 * v + 1] = b1;
        }
        v16bf b = __builtin_bit_cast(v16bf, bu);

        #pragma unroll
        for (int t = 0; t < MTILES; ++t) {
            // ---- A fragment: 16x32 bf16, row = lane&15, K pairs per VGPR ----
            const unsigned short* ar = &As[(t * 16 + n) * 128];
            v16u16 au;
            #pragma unroll
            for (int j = 0; j < 4; ++j) {                // VGPR 0..3: K 0..15
                int k = kb + hh * 8 + 2 * j;
                au[2 * j]     = ar[k];
                au[2 * j + 1] = ar[k + 1];
            }
            #pragma unroll
            for (int j = 0; j < 4; ++j) {                // VGPR 4..7: K 16..31
                int k = kb + 16 + hh * 8 + 2 * j;
                au[8 + 2 * j] = ar[k];
                au[9 + 2 * j] = ar[k + 1];
            }
            v16bf a = __builtin_bit_cast(v16bf, au);
            c[t] = __builtin_amdgcn_wmma_f32_16x16x32_bf16(
                       /*neg_a=*/false, a, /*neg_b=*/false, b,
                       /*c_mod=*/(short)0, c[t], /*reuse_a=*/false, /*reuse_b=*/false);
        }
    }

    // ---- epilogue: BN affine (per output column) + ReLU, store ----
    float s = 1.0f, sh = 0.0f;
    if (DO_BN) {
        float iv = rsqrtf(var[ocol] + 1e-5f);
        s  = gamma[ocol] * iv;
        sh = beta[ocol] - mean[ocol] * s;
    }
    if (NCOLS == 40 && ocol >= GIN_C) return;   // masked column tile (40..47)

    #pragma unroll
    for (int t = 0; t < MTILES; ++t) {
        #pragma unroll
        for (int j = 0; j < 8; ++j) {
            int m = m0 + t * 16 + j + 8 * hh;   // C/D layout: M = vgpr + 8*half
            float val = c[t][j];
            if (DO_BN) val = val * s + sh;
            val = fmaxf(val, 0.0f);
            if (OUT_BF16) ((unsigned short*)Dout)[(size_t)m * NCOLS + ocol] = f2bf(val);
            else          ((float*)Dout)[(size_t)m * NCOLS + ocol] = val;
            if (DUAL_OUT) Dout2[(size_t)m * GIN_H + ocol] = val;   // agg2 init copy
        }
    }
}

// ---------------------------------------------------------------------------
extern "C" void kernel_launch(void* const* d_in, const int* in_sizes, int n_in,
                              void* d_out, int out_size, void* d_ws, size_t ws_size,
                              hipStream_t stream) {
    const float* x     = (const float*)d_in[0];
    const int*   ei    = (const int*)d_in[1];
    const float* W1a   = (const float*)d_in[2];
    const float* bng   = (const float*)d_in[3];
    const float* bnb   = (const float*)d_in[4];
    const float* bnm   = (const float*)d_in[5];
    const float* bnv   = (const float*)d_in[6];
    const float* W1b   = (const float*)d_in[7];
    const float* W2a   = (const float*)d_in[8];
    const float* W2b   = (const float*)d_in[9];
    float* out = (float*)d_out;

    // workspace layout (all offsets 256B-aligned)
    char* ws = (char*)d_ws;
    float*          buf0 = (float*)ws;                                  // 25.6 MB: agg1, later agg2
    unsigned short* buf1 = (unsigned short*)(ws + 25600000);            // 12.8 MB: h1 bf16, later h3
    float*          buf2 = (float*)(ws + 25600000 + 12800000);          // 25.6 MB: h2 fp32
    unsigned short* wb   = (unsigned short*)(ws + 25600000 + 12800000 + 25600000);
    unsigned short* w1a = wb;
    unsigned short* w1b = w1a + 128 * 128;
    unsigned short* w2a = w1b + 128 * 128;
    unsigned short* w2b = w2a + 128 * 128;

    const int MB = GIN_N / MROWS;                    // 625 exact M-blocks
    const int scatter_blocks = (GIN_E * 32) / 256;   // 1 wave per edge

    // weights -> bf16
    gin_f2bf<<<(128 * 128 + 255) / 256, 256, 0, stream>>>(W1a, w1a, 128 * 128);
    gin_f2bf<<<(128 * 128 + 255) / 256, 256, 0, stream>>>(W1b, w1b, 128 * 128);
    gin_f2bf<<<(128 * 128 + 255) / 256, 256, 0, stream>>>(W2a, w2a, 128 * 128);
    gin_f2bf<<<(GIN_C * 128 + 255) / 256, 256, 0, stream>>>(W2b, w2b, GIN_C * 128);

    // agg1 = x + sum_{j->i} x_j
    gin_copy4<<<(GIN_N * GIN_F / 4 + 255) / 256, 256, 0, stream>>>(
        (const float4*)x, (float4*)buf0, GIN_N * GIN_F / 4);
    gin_scatter<<<scatter_blocks, 256, 0, stream>>>(ei, x, buf0);

    // h1 = relu(BN(agg1 @ W1a^T))  -> bf16
    gin_gemm<false, true, true, false, 128><<<MB, 256, 0, stream>>>(
        buf0, w1a, bng, bnb, bnm, bnv, buf1, nullptr);

    // h2 = relu(h1 @ W1b^T) -> fp32 buf2, and dup into buf0 as agg2 init
    gin_gemm<true, false, false, true, 128><<<MB, 256, 0, stream>>>(
        buf1, w1b, nullptr, nullptr, nullptr, nullptr, buf2, buf0);

    // agg2 = h2 + sum_{j->i} h2_j
    gin_scatter<<<scatter_blocks, 256, 0, stream>>>(ei, buf2, buf0);

    // h3 = relu(agg2 @ W2a^T) -> bf16
    gin_gemm<false, false, true, false, 128><<<MB, 256, 0, stream>>>(
        buf0, w2a, nullptr, nullptr, nullptr, nullptr, buf1, nullptr);

    // out = relu(h3 @ W2b^T) -> fp32 [50000 x 40]
    gin_gemm<true, false, false, false, 40><<<MB, 96, 0, stream>>>(
        buf1, w2b, nullptr, nullptr, nullptr, nullptr, out, nullptr);
}
